// VersorLinear_23502061043877
// MI455X (gfx1250) — matrix operand order repaired and checked by reading, not compiled, not run on hardware
//
#include <hip/hip_runtime.h>

typedef __attribute__((ext_vector_type(16))) _Float16 v16h;
typedef __attribute__((ext_vector_type(8)))  _Float16 v8h;
typedef __attribute__((ext_vector_type(8)))  float    v8f;

#define IN_F   256
#define OUT_F  256
#define KDIM   (IN_F * 32)    // 8192
#define NDIM   (OUT_F * 32)   // 8192

#define BM 128
#define BN 128
#define BK 64

#define AS_STRIDE 72    // halves (64 + 8 pad)  -> 144 B row stride (16B aligned)
#define BT_STRIDE 72    // halves (64 + 8 pad)  -> K-major (transposed) B tile

// ---------------- compile-time Cl(4,1) Cayley sign table ----------------
// gp[j,l,k] has its single +/-1 entry at k = j^l; sign[l][k] = gp_sign(k^l, l).
struct SignBits { unsigned v[32]; };

constexpr int gp_sign(int a, int b) {
  int sign = 1, bits = a;
  for (int i = 0; i < 5; ++i) {
    if ((b >> i) & 1) {
      for (int j = i + 1; j < 5; ++j)
        if ((bits >> j) & 1) sign = -sign;
      if ((bits >> i) & 1) {
        if (i == 4) sign = -sign;   // e5^2 = -1 (signature 4,1)
        bits &= ~(1 << i);
      } else {
        bits |= (1 << i);
      }
    }
  }
  return sign;
}

constexpr SignBits make_signbits() {
  SignBits s{};
  for (int l = 0; l < 32; ++l) {
    unsigned m = 0;
    for (int k = 0; k < 32; ++k)
      if (gp_sign(k ^ l, l) < 0) m |= (1u << k);
    s.v[l] = m;
  }
  return s;
}

__constant__ SignBits SIGNBITS = make_signbits();

// ------------------------------- kernel --------------------------------
__global__ __launch_bounds__(256)
void versor_linear_wmma(const float* __restrict__ X,
                        const float* __restrict__ W,
                        float* __restrict__ O)
{
  // Staging LDS (37888 B), reused by the epilogue (33792 B).
  __shared__ __align__(16) unsigned char smem[37888];
  _Float16* As  = (_Float16*)smem;                         // [BM][AS_STRIDE]
  _Float16* Bs  = (_Float16*)(smem + BM * AS_STRIDE * 2);  // [BN][BT_STRIDE]  K-major!
  float*    Ws  = (float*)(smem + BM * AS_STRIDE * 2 + BN * BT_STRIDE * 2); // [8][32]
  float*    Epi = (float*)smem;                            // [8 waves][32][33]

  const int tid  = threadIdx.x;
  const int lane = tid & 31;
  const int wid  = tid >> 5;      // 0..7
  const int wm   = wid & 1;       // 64-row slab
  const int wn   = wid >> 1;      // 32-col slab (== one output feature's 32 blades)
  const int g    = lane >> 4;
  const int l16  = lane & 15;

  const int m0 = blockIdx.y * BM;
  const int n0 = blockIdx.x * BN;
  const int o0 = blockIdx.x * 4;  // 4 output features per 128-col tile

  v8f acc[4][2];
  #pragma unroll
  for (int a = 0; a < 4; ++a)
    #pragma unroll
    for (int b = 0; b < 2; ++b)
      #pragma unroll
      for (int e = 0; e < 8; ++e) acc[a][b][e] = 0.0f;

  const int c4 = tid & 15;   // float4 column within the 64-wide K slab
  const int r0 = tid >> 4;   // row group 0..15

  for (int kt = 0; kt < KDIM / BK; ++kt) {
    const int k0 = kt * BK;
    const int i0 = kt * 2;   // two input features per K-step

    __syncthreads();  // previous compute done before restaging

    // ---- stage x tile 128x64 (fp32 -> fp16), 8 float4 per thread ----
    float4 xr[8];
    #pragma unroll
    for (int q = 0; q < 8; ++q) {
      const int r = r0 + 16 * q;
      xr[q] = *(const float4*)(X + (size_t)(m0 + r) * KDIM + k0 + c4 * 4);
    }
    // ---- stage weight chunk: 8 x 32 fp32  (4 o_loc x 2 i x 32 blades) ----
    {
      const int o_loc = tid >> 6;
      const int tt    = (tid >> 5) & 1;
      const int j     = tid & 31;
      Ws[(o_loc * 2 + tt) * 32 + j] =
          W[(size_t)(o0 + o_loc) * KDIM + (size_t)(i0 + tt) * 32 + j];
    }
    #pragma unroll
    for (int q = 0; q < 8; ++q) {
      const int r = r0 + 16 * q;
      _Float16* dst = As + r * AS_STRIDE + c4 * 4;
      dst[0] = (_Float16)xr[q].x;
      dst[1] = (_Float16)xr[q].y;
      dst[2] = (_Float16)xr[q].z;
      dst[3] = (_Float16)xr[q].w;
    }
    __syncthreads();  // Ws visible

    // ---- expand Cayley-structured operator tile, K-major:
    //      Bs[col = o_loc*32+k][tt*32+l] = w[o,i,k^l] * sign[l][k]  (fp16)
    //      each thread owns one column-half -> 32 contiguous LDS halves ----
    {
      const int col   = tid & 127;     // o_loc*32 + k
      const int tt    = tid >> 7;      // 0..1  (which input feature of the pair)
      const int o_loc = col >> 5;
      const int k     = col & 31;
      const float* wrow = Ws + (o_loc * 2 + tt) * 32;
      _Float16* dst = Bs + col * BT_STRIDE + tt * 32;
      #pragma unroll
      for (int l = 0; l < 32; ++l) {
        float v = wrow[k ^ l];
        v = ((SIGNBITS.v[l] >> k) & 1u) ? -v : v;
        dst[l] = (_Float16)v;
      }
    }
    __syncthreads();  // As + Bs ready

    // ---- 2 x (K=32) WMMA substeps ----
    #pragma unroll
    for (int kk = 0; kk < BK; kk += 32) {
      // B fragments: lane needs K = kk+16g+0..15 at fixed col -> contiguous!
      v16h bfr[2];
      #pragma unroll
      for (int ns = 0; ns < 2; ++ns) {
        const _Float16* bp =
            Bs + (wn * 32 + ns * 16 + l16) * BT_STRIDE + kk + 16 * g;
        const v8h blo = *(const v8h*)bp;        // K = kk+16g + 0..7
        const v8h bhi = *(const v8h*)(bp + 8);  // K = kk+16g + 8..15
        #pragma unroll
        for (int e = 0; e < 8; ++e) {
          bfr[ns][e]     = blo[e];
          bfr[ns][8 + e] = bhi[e];
        }
      }
      #pragma unroll
      for (int ms = 0; ms < 4; ++ms) {
        // A fragment: lane needs K = kk+16g+0..15 -> 2x b128
        const _Float16* ap =
            As + (wm * 64 + ms * 16 + l16) * AS_STRIDE + kk + 16 * g;
        const v8h alo = *(const v8h*)ap;         // K = kk+16g + 0..7
        const v8h ahi = *(const v8h*)(ap + 8);   // K = kk+16g + 8..15
        v16h afr;
        #pragma unroll
        for (int e = 0; e < 8; ++e) {
          afr[e]     = alo[e];
          afr[8 + e] = ahi[e];
        }
        acc[ms][0] = __builtin_amdgcn_wmma_f32_16x16x32_f16(
            false, afr, false, bfr[0], (short)0, acc[ms][0], false, false);
        acc[ms][1] = __builtin_amdgcn_wmma_f32_16x16x32_f16(
            false, afr, false, bfr[1], (short)0, acc[ms][1], false, false);
      }
    }
  }

  // ---- fused Cl(4,1) normalization epilogue (wave-local: full 32-blade
  //      range per wave), two phases to fit in the staging LDS footprint ----
  float* myEpi = Epi + wid * (32 * 33);
  #pragma unroll
  for (int p = 0; p < 2; ++p) {
    __syncthreads();  // protect LDS reuse (staging / previous phase)
    #pragma unroll
    for (int msl = 0; msl < 2; ++msl) {
      const int ms = 2 * p + msl;
      #pragma unroll
      for (int ns = 0; ns < 2; ++ns)
        #pragma unroll
        for (int v = 0; v < 8; ++v)
          myEpi[(msl * 16 + g * 8 + v) * 33 + ns * 16 + l16] = acc[ms][ns][v];
    }
    __syncthreads();

    // each lane normalizes one full multivector row (32 blades)
    const float* row = myEpi + lane * 33;
    float vals[32];
    float ss = 0.0f, gs = 0.0f;
    #pragma unroll
    for (int k = 0; k < 32; ++k) {
      const float v = row[k];
      vals[k] = v;
      const float sq = v * v;
      ss += sq;
      const int grade = __popc(k);
      const int neg = (((grade * (grade - 1) / 2) & 1) ^ ((k >> 4) & 1));
      gs += neg ? -sq : sq;
    }
    const float g_norm = sqrtf(fabsf(gs) + 1e-8f);
    const float f_norm = sqrtf(ss) + 1e-8f;
    const float denom  = fmaxf(fmaxf(g_norm, f_norm * 0.25f), 1.0f);
    const float inv    = 1.0f / denom;

    const int mrow = m0 + wm * 64 + p * 32 + lane;
    float* orow = O + (size_t)mrow * NDIM + n0 + wn * 32;
    #pragma unroll
    for (int q = 0; q < 8; ++q) {
      float4 ov;
      ov.x = vals[4 * q + 0] * inv;
      ov.y = vals[4 * q + 1] * inv;
      ov.z = vals[4 * q + 2] * inv;
      ov.w = vals[4 * q + 3] * inv;
      *(float4*)(orow + 4 * q) = ov;
    }
  }
}

// ------------------------------ launcher -------------------------------
extern "C" void kernel_launch(void* const* d_in, const int* in_sizes, int n_in,
                              void* d_out, int out_size, void* d_ws, size_t ws_size,
                              hipStream_t stream) {
  const float* x = (const float*)d_in[0];   // [4,1024,256,32] fp32
  const float* w = (const float*)d_in[1];   // [256,256,32]    fp32
  float* out = (float*)d_out;               // [4,1024,256,32] fp32

  const int M = in_sizes[0] / KDIM;         // 4096
  dim3 grid(NDIM / BN, M / BM);             // (64, 32) = 2048 workgroups
  versor_linear_wmma<<<grid, dim3(256), 0, stream>>>(x, w, out);
}
